// Head_2929167695983
// MI455X (gfx1250) — matrix operand order repaired and checked by reading, not compiled, not run on hardware
//
#include <hip/hip_runtime.h>

// MI455X / gfx1250 fused causal-attention head.
//   B=4, T=4096, E=1024, H=64, fp32 in/out, f16 WMMA compute (f32 accumulate).
// Phase 0: weights [E,H] f32 -> [H,E] f16 (transposed so B-fragments are dword loads)
// Phase 1: QKV projection via v_wmma_f32_16x16x32_f16; Q,K -> [B*T,H] f16, V -> [B][H][T] f16
// Phase 2: flash attention (online softmax), one wave per 16-query tile, 32-key blocks.

typedef __attribute__((ext_vector_type(16))) _Float16 v16h;
typedef __attribute__((ext_vector_type(2)))  _Float16 v2h;
typedef __attribute__((ext_vector_type(8)))  float    v8f;
typedef __attribute__((ext_vector_type(2)))  float    v2f;

constexpr int Bk = 4;
constexpr int Tk = 4096;
constexpr int Ek = 1024;
constexpr int Hk = 64;

__device__ __forceinline__ v8f wmma16(v16h a, v16h b, v8f c) {
  // D = A(16x32 f16) x B(32x16 f16) + C(16x16 f32)
  return __builtin_amdgcn_wmma_f32_16x16x32_f16(
      /*neg_a=*/false, a, /*neg_b=*/false, b,
      /*c_mod=*/(short)0, c, /*reuse_a=*/false, /*reuse_b=*/false);
}

// B fragment (32x16): element (K = (j>>2)*16 + hf*8 + 2*(j&3) + {0,1}, N = col) at base[N*ldn + K].
// K pairs are contiguous in memory -> one b32 load per VGPR.
__device__ __forceinline__ v16h load_b_frag(const _Float16* __restrict__ base,
                                            int ldn, int lane) {
  const int col = lane & 15;
  const int hf  = lane >> 4;
  v16h b;
#pragma unroll
  for (int j = 0; j < 8; ++j) {
    v2h t = *(const v2h*)(base + (size_t)col * ldn + ((j >> 2) << 4) + (hf << 3) + ((j & 3) << 1));
    b[2 * j]     = t[0];
    b[2 * j + 1] = t[1];
  }
  return b;
}

// ---------------- Phase 0: weight transpose + f32->f16 ----------------
__global__ void __launch_bounds__(256) wprep_kernel(
    const float* __restrict__ Wk, const float* __restrict__ Wq,
    const float* __restrict__ Wv,
    _Float16* __restrict__ tK, _Float16* __restrict__ tQ,
    _Float16* __restrict__ tV) {
  int id = blockIdx.x * 256 + threadIdx.x;      // 3 * H * E threads
  int w  = id >> 16;                            // H*E = 65536
  int r  = id & 65535;
  int h  = r >> 10;                             // / E
  int e  = r & (Ek - 1);
  const float* src = (w == 0) ? Wk : (w == 1) ? Wq : Wv;
  _Float16*    dst = (w == 0) ? tK : (w == 1) ? tQ : tV;
  dst[(size_t)h * Ek + e] = (_Float16)src[(size_t)e * Hk + h];
}

// ---------------- Phase 1: fused QKV projection ----------------
__global__ void __launch_bounds__(256) qkv_proj_kernel(
    const float* __restrict__ x,
    const _Float16* __restrict__ tQ, const _Float16* __restrict__ tK,
    const _Float16* __restrict__ tV,
    _Float16* __restrict__ Qh, _Float16* __restrict__ Kh,
    _Float16* __restrict__ Vt) {
  const int lane = threadIdx.x & 31;
  const int wave = threadIdx.x >> 5;
  const int row0 = (blockIdx.x * 8 + wave) * 16;   // 16 rows of x per wave
  const int col  = lane & 15;
  const int hf   = lane >> 4;

  v8f accQ[4], accK[4], accV[4];
#pragma unroll
  for (int n = 0; n < 4; ++n) { accQ[n] = {}; accK[n] = {}; accV[n] = {}; }

  for (int k0 = 0; k0 < Ek; k0 += 32) {
    // A fragment (16x32) from fp32 x, converted to f16 in registers.
    v16h a;
#pragma unroll
    for (int i = 0; i < 8; ++i) {
      int k = ((i >> 2) << 4) + (hf << 3) + ((i & 3) << 1);
      v2f t = *(const v2f*)(x + (size_t)(row0 + col) * Ek + k0 + k);
      a[2 * i]     = (_Float16)t[0];
      a[2 * i + 1] = (_Float16)t[1];
    }
#pragma unroll
    for (int n = 0; n < 4; ++n) {
      v16h bq = load_b_frag(tQ + (size_t)(n * 16) * Ek + k0, Ek, lane);
      accQ[n] = wmma16(a, bq, accQ[n]);
      v16h bk = load_b_frag(tK + (size_t)(n * 16) * Ek + k0, Ek, lane);
      accK[n] = wmma16(a, bk, accK[n]);
      v16h bv = load_b_frag(tV + (size_t)(n * 16) * Ek + k0, Ek, lane);
      accV[n] = wmma16(a, bv, accV[n]);
    }
  }

  // D layout: VGPR r, lane -> (M = r + hf*8, N = col)
#pragma unroll
  for (int n = 0; n < 4; ++n) {
#pragma unroll
    for (int r = 0; r < 8; ++r) {
      int g = row0 + r + hf * 8;               // global row in [0, B*T)
      Qh[(size_t)g * Hk + n * 16 + col] = (_Float16)accQ[n][r];
      Kh[(size_t)g * Hk + n * 16 + col] = (_Float16)accK[n][r];
      int bb = g >> 12;                        // / T
      int tt = g & (Tk - 1);
      Vt[((size_t)bb * Hk + n * 16 + col) * Tk + tt] = (_Float16)accV[n][r];
    }
  }
}

// ---------------- Phase 2: flash attention ----------------
__global__ void __launch_bounds__(256) attn_kernel(
    const _Float16* __restrict__ Qh, const _Float16* __restrict__ Kh,
    const _Float16* __restrict__ Vt, float* __restrict__ out) {
  __shared__ _Float16 ldsP[8][16 * 32];        // per-wave P staging (C->A transpose)

  const int lane = threadIdx.x & 31;
  const int wave = threadIdx.x >> 5;
  const int qt   = blockIdx.x * 8 + wave;      // query tile id, 0..1023
  const int bb   = qt >> 8;                    // T/16 = 256 tiles per batch
  const int q0   = (qt & 255) * 16;
  const int col  = lane & 15;
  const int hf   = lane >> 4;

  const _Float16* Qb = Qh + ((size_t)bb * Tk + q0) * Hk;
  const _Float16* Kb = Kh + (size_t)bb * Tk * Hk;
  const _Float16* Vb = Vt + (size_t)bb * Hk * Tk;

  // Q A-fragments for h0 = 0 and 32 (resident across the key loop).
  v16h aq[2];
#pragma unroll
  for (int hh = 0; hh < 2; ++hh) {
#pragma unroll
    for (int i = 0; i < 8; ++i) {
      int k = hh * 32 + ((i >> 2) << 4) + (hf << 3) + ((i & 3) << 1);
      v2h t = *(const v2h*)(Qb + (size_t)col * Hk + k);
      aq[hh][2 * i]     = t[0];
      aq[hh][2 * i + 1] = t[1];
    }
  }

  v8f O[4];
#pragma unroll
  for (int n = 0; n < 4; ++n) O[n] = {};
  float m[8], l[8];
#pragma unroll
  for (int r = 0; r < 8; ++r) { m[r] = -__builtin_inff(); l[r] = 0.f; }

  const float SCALE = 0.03125f;                // 1/sqrt(E) = 1/32

  for (int key0 = 0; key0 < q0 + 16; key0 += 32) {
    if (key0 + 32 < q0 + 16) {                 // warm next K block (global_prefetch_b8)
      __builtin_prefetch(Kb + (size_t)(key0 + 32 + col) * Hk + hf * 32, 0, 0);
    }

    // S = q . k^T : two 16-key subtiles, K-dim = H in two chunks of 32.
    v8f S0 = {}, S1 = {};
#pragma unroll
    for (int hh = 0; hh < 2; ++hh) {
      v16h b0 = load_b_frag(Kb + (size_t)key0 * Hk + hh * 32, Hk, lane);
      v16h b1 = load_b_frag(Kb + (size_t)(key0 + 16) * Hk + hh * 32, Hk, lane);
      S0 = wmma16(aq[hh], b0, S0);
      S1 = wmma16(aq[hh], b1, S1);
    }

    // scale + causal mask + online softmax (row = r + hf*8, key = key0 [+16] + col)
    float corr[8];
#pragma unroll
    for (int r = 0; r < 8; ++r) {
      int   qrow = q0 + r + hf * 8;
      float s0 = (key0 + col      > qrow) ? -__builtin_inff() : S0[r] * SCALE;
      float s1 = (key0 + 16 + col > qrow) ? -__builtin_inff() : S1[r] * SCALE;
      float t = fmaxf(s0, s1);                 // row-max across 16-lane half group
      t = fmaxf(t, __shfl_xor(t, 1));
      t = fmaxf(t, __shfl_xor(t, 2));
      t = fmaxf(t, __shfl_xor(t, 4));
      t = fmaxf(t, __shfl_xor(t, 8));
      float nm = fmaxf(m[r], t);
      corr[r] = __expf(m[r] - nm);
      m[r] = nm;
      float p0 = __expf(s0 - nm);
      float p1 = __expf(s1 - nm);
      S0[r] = p0;
      S1[r] = p1;
      float rs = p0 + p1;                      // row-sum across half group
      rs += __shfl_xor(rs, 1);
      rs += __shfl_xor(rs, 2);
      rs += __shfl_xor(rs, 4);
      rs += __shfl_xor(rs, 8);
      l[r] = l[r] * corr[r] + rs;
    }

    // P: C-layout registers -> LDS [16][32] -> A-layout registers (cross-lane transpose)
#pragma unroll
    for (int r = 0; r < 8; ++r) {
      int mrow = r + hf * 8;
      ldsP[wave][mrow * 32 + col]      = (_Float16)S0[r];
      ldsP[wave][mrow * 32 + 16 + col] = (_Float16)S1[r];
    }
    asm volatile("s_wait_dscnt 0" ::: "memory"); // same-wave LDS RAW fence
    v16h aP;
#pragma unroll
    for (int i = 0; i < 8; ++i) {
      int k = ((i >> 2) << 4) + (hf << 3) + ((i & 3) << 1);
      v2h t = *(const v2h*)(&ldsP[wave][col * 32 + k]);
      aP[2 * i]     = t[0];
      aP[2 * i + 1] = t[1];
    }

    // O = O*corr + P @ V  (V stored [H][T] so B K-pairs are contiguous dwords)
#pragma unroll
    for (int n = 0; n < 4; ++n) {
#pragma unroll
      for (int r = 0; r < 8; ++r) O[n][r] *= corr[r];
      v16h bv = load_b_frag(Vb + (size_t)(n * 16) * Tk + key0, Tk, lane);
      O[n] = wmma16(aP, bv, O[n]);
    }
  }

  // out[b, q0+M, h] = O / l
#pragma unroll
  for (int r = 0; r < 8; ++r) {
    float inv = 1.0f / l[r];
    size_t g = (size_t)bb * Tk + q0 + r + hf * 8;
#pragma unroll
    for (int n = 0; n < 4; ++n)
      out[g * Hk + n * 16 + col] = O[n][r] * inv;
  }
}

// ---------------- launcher ----------------
extern "C" void kernel_launch(void* const* d_in, const int* in_sizes, int n_in,
                              void* d_out, int out_size, void* d_ws, size_t ws_size,
                              hipStream_t stream) {
  const float* x  = (const float*)d_in[0];
  const float* Wk = (const float*)d_in[1];
  const float* Wq = (const float*)d_in[2];
  const float* Wv = (const float*)d_in[3];
  float* out = (float*)d_out;

  // Workspace layout (f16): 3 transposed weights + Q + K + V^T  (~6.7 MB)
  _Float16* ws = (_Float16*)d_ws;
  _Float16* tK = ws;
  _Float16* tQ = tK + (size_t)Hk * Ek;
  _Float16* tV = tQ + (size_t)Hk * Ek;
  _Float16* Qh = tV + (size_t)Hk * Ek;
  _Float16* Kh = Qh + (size_t)Bk * Tk * Hk;
  _Float16* Vt = Kh + (size_t)Bk * Tk * Hk;

  wprep_kernel<<<(3 * Hk * Ek) / 256, 256, 0, stream>>>(Wk, Wq, Wv, tK, tQ, tV);
  qkv_proj_kernel<<<(Bk * Tk) / (16 * 8), 256, 0, stream>>>(x, tQ, tK, tV, Qh, Kh, Vt);
  attn_kernel<<<(Bk * Tk) / (16 * 8), 256, 0, stream>>>(Qh, Kh, Vt, out);
}
